// LSTM_cell_AR_77549929496853
// MI455X (gfx1250) — matrix operand order repaired and checked by reading, not compiled, not run on hardware
//
#include <hip/hip_runtime.h>
#include <hip/hip_bf16.h>

// ---------------------------------------------------------------------------
// Problem constants (from reference)
// ---------------------------------------------------------------------------
#define BB     256      // batch
#define TT     256      // total time
#define WIN    128      // window
#define II     8        // input features
#define HH     512      // hidden
#define G4     2048     // 4*H
#define OSTR   129      // output columns (1 + 128)

typedef __attribute__((ext_vector_type(16))) _Float16 v16h;
typedef __attribute__((ext_vector_type(8)))  _Float16 v8h;
typedef __attribute__((ext_vector_type(8)))  float    v8f;
typedef __attribute__((ext_vector_type(4)))  int      v4i;

// LDS row stride in halves: 520*2 = 1040 B = 260 dwords -> 4-bank skew per row,
// 16 rows * 4 banks * 4 dwords = all 64 banks, conflict-free b128 LDS loads.
#define LDK 520

// Address-space-qualified pointer types for the async global->LDS builtin
typedef __attribute__((address_space(1))) v4i* gv4ip;
typedef __attribute__((address_space(3))) v4i* lv4ip;

__device__ inline void async_copy_b128(const void* gsrc, void* ldst) {
    __builtin_amdgcn_global_load_async_to_lds_b128((gv4ip)gsrc, (lv4ip)ldst, 0, 0);
}

// ---------------------------------------------------------------------------
// Small prep kernels
// ---------------------------------------------------------------------------
__global__ void k_f32_to_f16(const float* __restrict__ src,
                             _Float16* __restrict__ dst, int n) {
    int i = blockIdx.x * blockDim.x + threadIdx.x;
    if (i < n) dst[i] = (_Float16)src[i];
}

// pad each row of (rows x K) to (rows x Kp) with zeros, converting to f16
__global__ void k_f32_to_f16_pad(const float* __restrict__ src,
                                 _Float16* __restrict__ dst,
                                 int rows, int K, int Kp) {
    int i = blockIdx.x * blockDim.x + threadIdx.x;
    if (i >= rows * Kp) return;
    int r = i / Kp, c = i % Kp;
    dst[i] = (c < K) ? (_Float16)src[r * K + c] : (_Float16)0.0f;
}

__global__ void k_bias_sum(const float* __restrict__ a,
                           const float* __restrict__ b,
                           float* __restrict__ o, int n) {
    int i = blockIdx.x * blockDim.x + threadIdx.x;
    if (i < n) o[i] = a[i] + b[i];
}

__global__ void k_zero_f32(float* p, int n) {
    int i = blockIdx.x * blockDim.x + threadIdx.x;
    if (i < n) p[i] = 0.0f;
}
__global__ void k_zero_f16(_Float16* p, int n) {
    int i = blockIdx.x * blockDim.x + threadIdx.x;
    if (i < n) p[i] = (_Float16)0.0f;
}

// AR feature: feat[b,0] = out[b, prevcol]; feat[b,1..7] = x[b, t, 1..7]
__global__ void k_feat_build(const float* __restrict__ x,
                             const float* __restrict__ out,
                             float* __restrict__ feat, int t, int prevcol) {
    int i = blockIdx.x * blockDim.x + threadIdx.x;
    if (i >= BB * II) return;
    int b = i >> 3, c = i & 7;
    feat[i] = (c == 0) ? out[b * OSTR + prevcol]
                       : x[(size_t)b * (TT * II) + t * II + c];
}

// out[:, col] = h @ linW.T + linb   (one wave per batch row)
__global__ void k_linear_out(const _Float16* __restrict__ h,
                             const float* __restrict__ linW,
                             const float* __restrict__ linb,
                             float* __restrict__ out, int col) {
    int wv = threadIdx.x >> 5, lane = threadIdx.x & 31;
    int b = blockIdx.x * 8 + wv;
    float s = 0.0f;
    for (int k = lane; k < HH; k += 32)
        s += (float)h[(size_t)b * HH + k] * linW[k];
#pragma unroll
    for (int off = 16; off > 0; off >>= 1) s += __shfl_xor(s, off, 32);
    if (lane == 0) out[b * OSTR + col] = s + linb[0];
}

// ---------------------------------------------------------------------------
// WMMA fragment loaders (gfx1250 wave32 layouts, 16x16x32 f16 -> f32)
// A 16x32: lane = M row (lane&15); half i -> K = (i>=8?16:0) + hs*8 + (i&7)
// B 32x16: lane = N col (lane&15); half i -> K = (i>=8?16:0) + hs*8 + (i&7)
// ---------------------------------------------------------------------------
__device__ inline v16h frag_a(const _Float16* __restrict__ row, int hs, int k0) {
    const _Float16* p = row + k0 + hs * 8;
    v8h lo = *(const v8h*)p;
    v8h hi = *(const v8h*)(p + 16);
    v16h a;
#pragma unroll
    for (int i = 0; i < 8; ++i) { a[i] = lo[i]; a[i + 8] = hi[i]; }
    return a;
}
__device__ inline v16h frag_b(const _Float16* __restrict__ row, int hs, int k0) {
    const _Float16* p = row + k0 + hs * 8;
    v8h lo = *(const v8h*)p;
    v8h hi = *(const v8h*)(p + 16);
    v16h b;
#pragma unroll
    for (int i = 0; i < 8; ++i) { b[i] = lo[i]; b[i + 8] = hi[i]; }
    return b;
}

__device__ inline float sigm(float x) { return 1.0f / (1.0f + __expf(-x)); }

// ---------------------------------------------------------------------------
// Fused LSTM step:  gates = xin @ Wih.T + hprev @ Whh.T + bsum ;
// cell update ; hnext (f16) ; cstate in-place (f32).
// Grid: (B/64, H/16). Block: 512 threads = 16 waves = 4 m-subtiles x 4 gates.
// ---------------------------------------------------------------------------
__global__ __launch_bounds__(512)
void k_lstm_step(const _Float16* __restrict__ xin16,  // Kx==512 path (f16, ld 512)
                 const float*    __restrict__ xin32,  // Kx==32 path (f32, xStride, 8 valid cols)
                 int xStride, int Kx,
                 const _Float16* __restrict__ Wih,    // G4 x Kx f16
                 const _Float16* __restrict__ Whh,    // G4 x 512 f16
                 const float*    __restrict__ bsum,   // G4
                 const _Float16* __restrict__ hprev,  // B x 512 f16
                 float*          __restrict__ cstate, // B x 512 f32
                 _Float16*       __restrict__ hnext)  // B x 512 f16
{
    __shared__ _Float16 Ah[64][LDK];
    __shared__ _Float16 Xh[64][LDK];
    __shared__ float    Gts[4][64][16];

    const int tid    = threadIdx.x;
    const int lane   = tid & 31;
    const int wave   = tid >> 5;
    const int gate   = wave & 3;
    const int msub   = wave >> 2;
    const int hs     = lane >> 4;
    const int ln     = lane & 15;
    const int m0base = blockIdx.x * 64;
    const int n0     = blockIdx.y * 16;

    // ---- stage hprev tile (64 x 512 f16) into LDS via async DMA -----------
    for (int i = tid; i < 64 * 64; i += 512) {
        int r = i >> 6, cg = i & 63;
        async_copy_b128(hprev + (size_t)(m0base + r) * HH + cg * 8,
                        &Ah[r][cg * 8]);
    }
    // ---- stage x tile ------------------------------------------------------
    if (Kx == 512) {
        for (int i = tid; i < 64 * 64; i += 512) {
            int r = i >> 6, cg = i & 63;
            async_copy_b128(xin16 + (size_t)(m0base + r) * HH + cg * 8,
                            &Xh[r][cg * 8]);
        }
    } else {  // Kx == 32, 8 real input columns, rest zero (f32 -> f16 convert)
        for (int i = tid; i < 64 * 32; i += 512) {
            int r = i >> 5, c = i & 31;
            float v = (c < II) ? xin32[(size_t)(m0base + r) * xStride + c] : 0.0f;
            Xh[r][c] = (_Float16)v;
        }
    }
    __builtin_amdgcn_s_wait_asynccnt(0);
    __syncthreads();

    // ---- gate GEMM: 16x16 tile of gate `gate` for rows msub*16.. ----------
    const int j = gate * HH + n0 + ln;       // gate-unit row in W (N index)
    const float bb = bsum[j];
    v8f acc;
#pragma unroll
    for (int v = 0; v < 8; ++v) acc[v] = bb;

    const _Float16* arow = &Ah[msub * 16 + ln][0];
    const _Float16* xrow = &Xh[msub * 16 + ln][0];
    const _Float16* wh   = Whh + (size_t)j * HH;
    const _Float16* wi   = Wih + (size_t)j * Kx;

#pragma unroll 4
    for (int k0 = 0; k0 < HH; k0 += 32) {
        acc = __builtin_amdgcn_wmma_f32_16x16x32_f16(
            false, frag_a(arow, hs, k0), false, frag_b(wh, hs, k0),
            (short)0, acc, false, false);
    }
#pragma unroll 4
    for (int k0 = 0; k0 < Kx; k0 += 32) {
        acc = __builtin_amdgcn_wmma_f32_16x16x32_f16(
            false, frag_a(xrow, hs, k0), false, frag_b(wi, hs, k0),
            (short)0, acc, false, false);
    }

    // ---- scatter gate tile to LDS -----------------------------------------
#pragma unroll
    for (int v = 0; v < 8; ++v)
        Gts[gate][msub * 16 + hs * 8 + v][ln] = acc[v];
    __syncthreads();

    // ---- fused cell update -------------------------------------------------
    for (int e = tid; e < 64 * 16; e += 512) {
        int r = e >> 4, c = e & 15;
        int b = m0base + r;
        int hj = n0 + c;
        float ig = Gts[0][r][c];
        float fg = Gts[1][r][c];
        float gg = Gts[2][r][c];
        float og = Gts[3][r][c];
        size_t idx = (size_t)b * HH + hj;
        float c_old = cstate[idx];
        float c_new = sigm(fg) * c_old + sigm(ig) * tanhf(gg);
        float h_new = sigm(og) * tanhf(c_new);
        cstate[idx] = c_new;
        hnext[idx]  = (_Float16)h_new;
    }
}

// ---------------------------------------------------------------------------
// Host-side orchestration
// ---------------------------------------------------------------------------
extern "C" void kernel_launch(void* const* d_in, const int* in_sizes, int n_in,
                              void* d_out, int out_size, void* d_ws, size_t ws_size,
                              hipStream_t stream) {
    (void)in_sizes; (void)n_in; (void)out_size; (void)ws_size;

    const float* x    = (const float*)d_in[0];
    const float* Wih0 = (const float*)d_in[1];
    const float* Whh0 = (const float*)d_in[2];
    const float* bih0 = (const float*)d_in[3];
    const float* bhh0 = (const float*)d_in[4];
    const float* Wih1 = (const float*)d_in[5];
    const float* Whh1 = (const float*)d_in[6];
    const float* bih1 = (const float*)d_in[7];
    const float* bhh1 = (const float*)d_in[8];
    const float* cWih = (const float*)d_in[9];
    const float* cWhh = (const float*)d_in[10];
    const float* cbih = (const float*)d_in[11];
    const float* cbhh = (const float*)d_in[12];
    const float* linW = (const float*)d_in[13];
    const float* linb = (const float*)d_in[14];
    float* out = (float*)d_out;

    // ---- workspace carve ---------------------------------------------------
    char* w = (char*)d_ws;
    size_t off = 0;
    auto carve = [&](size_t bytes) -> char* {
        char* p = w + off;
        off = (off + bytes + 255) & ~(size_t)255;
        return p;
    };
    _Float16* Whh0h = (_Float16*)carve((size_t)G4 * HH * 2);
    _Float16* Wih1h = (_Float16*)carve((size_t)G4 * HH * 2);
    _Float16* Whh1h = (_Float16*)carve((size_t)G4 * HH * 2);
    _Float16* cWhhh = (_Float16*)carve((size_t)G4 * HH * 2);
    _Float16* Wih0p = (_Float16*)carve((size_t)G4 * 32 * 2);
    _Float16* cWihp = (_Float16*)carve((size_t)G4 * 32 * 2);
    float*    bsum0 = (float*)carve((size_t)G4 * 4);
    float*    bsum1 = (float*)carve((size_t)G4 * 4);
    float*    bsumc = (float*)carve((size_t)G4 * 4);
    _Float16* h0a   = (_Float16*)carve((size_t)BB * HH * 2);
    _Float16* h0b   = (_Float16*)carve((size_t)BB * HH * 2);
    float*    c0    = (float*)carve((size_t)BB * HH * 4);
    _Float16* h1a   = (_Float16*)carve((size_t)BB * HH * 2);
    _Float16* h1b   = (_Float16*)carve((size_t)BB * HH * 2);
    float*    c1    = (float*)carve((size_t)BB * HH * 4);
    float*    feat  = (float*)carve((size_t)BB * II * 4);

    // ---- prep --------------------------------------------------------------
    const int NWB = G4 * HH;  // 1M elements per big weight
    k_f32_to_f16<<<(NWB + 255) / 256, 256, 0, stream>>>(Whh0, Whh0h, NWB);
    k_f32_to_f16<<<(NWB + 255) / 256, 256, 0, stream>>>(Wih1, Wih1h, NWB);
    k_f32_to_f16<<<(NWB + 255) / 256, 256, 0, stream>>>(Whh1, Whh1h, NWB);
    k_f32_to_f16<<<(NWB + 255) / 256, 256, 0, stream>>>(cWhh, cWhhh, NWB);
    k_f32_to_f16_pad<<<(G4 * 32 + 255) / 256, 256, 0, stream>>>(Wih0, Wih0p, G4, II, 32);
    k_f32_to_f16_pad<<<(G4 * 32 + 255) / 256, 256, 0, stream>>>(cWih, cWihp, G4, II, 32);
    k_bias_sum<<<(G4 + 255) / 256, 256, 0, stream>>>(bih0, bhh0, bsum0, G4);
    k_bias_sum<<<(G4 + 255) / 256, 256, 0, stream>>>(bih1, bhh1, bsum1, G4);
    k_bias_sum<<<(G4 + 255) / 256, 256, 0, stream>>>(cbih, cbhh, bsumc, G4);
    const int NS = BB * HH;
    k_zero_f16<<<(NS + 255) / 256, 256, 0, stream>>>(h0a, NS);
    k_zero_f32<<<(NS + 255) / 256, 256, 0, stream>>>(c0, NS);
    k_zero_f16<<<(NS + 255) / 256, 256, 0, stream>>>(h1a, NS);
    k_zero_f32<<<(NS + 255) / 256, 256, 0, stream>>>(c1, NS);

    dim3 sgrid(BB / 64, HH / 16);   // (4, 32)
    dim3 sblk(512);

    // ---- window: layer0 + layer1, interleaved per step ---------------------
    _Float16 *h0c = h0a, *h0n = h0b;
    _Float16 *h1c = h1a, *h1n = h1b;
    for (int t = 0; t < WIN; ++t) {
        // layer0: x[:, t, :] (K=8 padded to 32)
        k_lstm_step<<<sgrid, sblk, 0, stream>>>(
            nullptr, x + (size_t)t * II, TT * II, 32,
            Wih0p, Whh0h, bsum0, h0c, c0, h0n);
        // layer1: input = fresh h0 (K=512)
        k_lstm_step<<<sgrid, sblk, 0, stream>>>(
            h0n, nullptr, 0, 512,
            Wih1h, Whh1h, bsum1, h1c, c1, h1n);
        { _Float16* tmp = h0c; h0c = h0n; h0n = tmp; }
        { _Float16* tmp = h1c; h1c = h1n; h1n = tmp; }
    }

    // out0 = h1_final @ linW.T + linb  -> column 0
    k_linear_out<<<32, 256, 0, stream>>>(h1c, linW, linb, out, 0);

    // ---- autoregressive phase ---------------------------------------------
    for (int s = 0; s < TT - WIN; ++s) {
        int t = WIN + s;
        k_feat_build<<<(BB * II + 255) / 256, 256, 0, stream>>>(x, out, feat, t, s);
        k_lstm_step<<<sgrid, sblk, 0, stream>>>(
            nullptr, feat, II, 32,
            cWihp, cWhhh, bsumc, h1c, c1, h1n);
        k_linear_out<<<32, 256, 0, stream>>>(h1n, linW, linb, out, s + 1);
        { _Float16* tmp = h1c; h1c = h1n; h1n = tmp; }
    }
}
